// MultiHeadAttention_25666724561197
// MI455X (gfx1250) — compile-verified, matrix operand order checked by reading, not run on hardware
//
#include <hip/hip_runtime.h>
#include <hip/hip_bf16.h>

// MHA forward for B=2,S=2048,D=1024,H=16,HD=64 on gfx1250 (wave32, WMMA).
// Pipeline: [qkv_proj f32->f16 + fold scale into Q] -> [flash attention, f16
// WMMA, online softmax] -> [out proj, f32 out + bias].
// All GEMM-ish loops are double-buffered in LDS; pure-copy stagings use
// GLOBAL_LOAD_ASYNC_TO_LDS_B128 (ASYNCcnt) when the builtin is available.

typedef __attribute__((ext_vector_type(16))) _Float16 v16h;
typedef __attribute__((ext_vector_type(8)))  _Float16 v8h;
typedef __attribute__((ext_vector_type(8)))  float    v8f;
typedef __attribute__((ext_vector_type(4)))  int      v4i;

#define B_   2
#define S_   2048
#define D_   1024
#define H_   16
#define HD_  64
#define SCALE_ 0.125f   // HD^-0.5

#if defined(__has_builtin)
#if __has_builtin(__builtin_amdgcn_global_load_async_to_lds_b128) && \
    __has_builtin(__builtin_amdgcn_s_wait_asynccnt)
#define HAVE_ASYNC_LDS 1
#endif
#endif

#ifdef HAVE_ASYNC_LDS
typedef __attribute__((address_space(1))) v4i gv4i_t;   // global int4
typedef __attribute__((address_space(3))) v4i lv4i_t;   // LDS int4
#endif

// 16-byte global->LDS copy: async (ASYNCcnt-tracked, no VGPR data) if the
// toolchain exposes the gfx1250 builtin, else a plain register copy.
static __device__ inline void async_cp16(const _Float16* g, _Float16* l) {
#ifdef HAVE_ASYNC_LDS
  __builtin_amdgcn_global_load_async_to_lds_b128(
      (gv4i_t*)(void*)(uintptr_t)(const void*)g,
      (lv4i_t*)(void*)(uintptr_t)(void*)l, 0, 0);
#else
  *(uint4*)l = *(const uint4*)g;
#endif
}
static __device__ inline void async_join() {
#ifdef HAVE_ASYNC_LDS
  __builtin_amdgcn_s_wait_asynccnt(0);
#endif
}

static __device__ inline v8f zerov8f() {
  v8f r = {0.f,0.f,0.f,0.f,0.f,0.f,0.f,0.f};
  return r;
}

// Per-lane A/B fragment slice: halves [koff..koff+7] and [koff+16..koff+23]
// of one LDS tile row (two ds_load_b128).
static __device__ inline v16h ldfrag(const _Float16* rowbase, int koff) {
  v8h lo = *(const v8h*)(rowbase + koff);
  v8h hi = *(const v8h*)(rowbase + koff + 16);
  return __builtin_shufflevector(lo, hi, 0,1,2,3,4,5,6,7,8,9,10,11,12,13,14,15);
}

static __device__ inline v8f wmma_f16(v16h a, v16h b, v8f c) {
  return __builtin_amdgcn_wmma_f32_16x16x32_f16(false, a, false, b, (short)0, c,
                                                false, false);
}

static __device__ inline v8h cvt2h(float4 a, float4 b) {
  v8h r;
  r[0]=(_Float16)a.x; r[1]=(_Float16)a.y; r[2]=(_Float16)a.z; r[3]=(_Float16)a.w;
  r[4]=(_Float16)b.x; r[5]=(_Float16)b.y; r[6]=(_Float16)b.z; r[7]=(_Float16)b.w;
  return r;
}

// ---------------------------------------------------------------------------
// Kernel 1: Y = X @ W^T + bias, written f16 head-major [B,H,S,HD].
// blockIdx.z selects 0->Q (scaled), 1->K, 2->V. Tile 128x64, BK=32, 8 waves.
// Double-buffered LDS; staging converts f32->f16 through registers.
// ---------------------------------------------------------------------------
__global__ __launch_bounds__(256)
void qkv_proj_kernel(const float* __restrict__ xq, const float* __restrict__ xk,
                     const float* __restrict__ xv,
                     const float* __restrict__ Wq, const float* __restrict__ Wk,
                     const float* __restrict__ Wv,
                     const float* __restrict__ bq, const float* __restrict__ bk,
                     const float* __restrict__ bv,
                     _Float16* __restrict__ Qh, _Float16* __restrict__ Kh,
                     _Float16* __restrict__ Vh) {
  const int z = blockIdx.z;
  const float* X    = (z == 0) ? xq : (z == 1) ? xk : xv;
  const float* W    = (z == 0) ? Wq : (z == 1) ? Wk : Wv;
  const float* bias = (z == 0) ? bq : (z == 1) ? bk : bv;
  _Float16*    out  = (z == 0) ? Qh : (z == 1) ? Kh : Vh;
  const float scale = (z == 0) ? SCALE_ : 1.0f;

  const int m0 = blockIdx.y * 128;
  const int n0 = blockIdx.x * 64;
  const int tid = threadIdx.x;
  const int w = tid >> 5, L = tid & 31;

  __shared__ alignas(16) _Float16 As[2][128][32];
  __shared__ alignas(16) _Float16 Ws[2][64][32];

  v8f acc[4];
#pragma unroll
  for (int nt = 0; nt < 4; ++nt) acc[nt] = zerov8f();

  const int ar = tid >> 1, ac = (tid & 1) * 16;   // A staging: 16 halves/thread
  const int wr = tid >> 2, wc = (tid & 3) * 8;    // W staging:  8 halves/thread
  const int koff = (L >> 4) * 8;
  const int arow = 16 * w + (L & 15);
  const int cin = L & 15;

  const float* aptr = X + (size_t)(m0 + ar) * D_ + ac;
  const float* wptr = W + (size_t)(n0 + wr) * D_ + wc;

  float4 ra0, ra1, ra2, ra3, rw0, rw1;
  ra0 = ((const float4*)aptr)[0]; ra1 = ((const float4*)aptr)[1];
  ra2 = ((const float4*)aptr)[2]; ra3 = ((const float4*)aptr)[3];
  rw0 = ((const float4*)wptr)[0]; rw1 = ((const float4*)wptr)[1];
  *(v8h*)&As[0][ar][ac]     = cvt2h(ra0, ra1);
  *(v8h*)&As[0][ar][ac + 8] = cvt2h(ra2, ra3);
  *(v8h*)&Ws[0][wr][wc]     = cvt2h(rw0, rw1);
  __syncthreads();

  const int NIT = D_ / 32;
  for (int it = 0; it < NIT; ++it) {
    const int cur = it & 1;
    if (it < NIT - 1) {  // issue next slab's global loads before compute
      const float* ap = aptr + (it + 1) * 32;
      const float* wp = wptr + (it + 1) * 32;
      ra0 = ((const float4*)ap)[0]; ra1 = ((const float4*)ap)[1];
      ra2 = ((const float4*)ap)[2]; ra3 = ((const float4*)ap)[3];
      rw0 = ((const float4*)wp)[0]; rw1 = ((const float4*)wp)[1];
      if (it < NIT - 2) {
        __builtin_prefetch(ap + 32, 0, 1);
        __builtin_prefetch(wp + 32, 0, 1);
      }
    }

    v16h a  = ldfrag(&As[cur][arow][0], koff);
    v16h b0 = ldfrag(&Ws[cur][cin][0], koff);
    v16h b1 = ldfrag(&Ws[cur][16 + cin][0], koff);
    v16h b2 = ldfrag(&Ws[cur][32 + cin][0], koff);
    v16h b3 = ldfrag(&Ws[cur][48 + cin][0], koff);
    acc[0] = wmma_f16(a, b0, acc[0]);
    acc[1] = wmma_f16(a, b1, acc[1]);
    acc[2] = wmma_f16(a, b2, acc[2]);
    acc[3] = wmma_f16(a, b3, acc[3]);

    if (it < NIT - 1) {  // stage next slab into the other buffer
      *(v8h*)&As[cur ^ 1][ar][ac]     = cvt2h(ra0, ra1);
      *(v8h*)&As[cur ^ 1][ar][ac + 8] = cvt2h(ra2, ra3);
      *(v8h*)&Ws[cur ^ 1][wr][wc]     = cvt2h(rw0, rw1);
    }
    __syncthreads();
  }

  // Epilogue: C layout -> head-major f16 [B,H,S,HD], fold bias (+ scale on Q).
  const int rbase = 16 * w + ((L >> 4) * 8);
#pragma unroll
  for (int nt = 0; nt < 4; ++nt) {
    const int n = n0 + nt * 16 + cin;
    const int hh = n >> 6, hd = n & 63;
    const float bn = bias[n];
#pragma unroll
    for (int vv = 0; vv < 8; ++vv) {
      const int m = m0 + rbase + vv;
      const int bb = m >> 11;          // / S_
      const int ss = m & (S_ - 1);
      const size_t off = (((size_t)bb * H_ + hh) * S_ + ss) * HD_ + hd;
      out[off] = (_Float16)((acc[nt][vv] + bn) * scale);
    }
  }
}

// ---------------------------------------------------------------------------
// Kernel 2: flash attention. Block = (b,h,128-query tile), 8 waves, each wave
// owns 16 query rows. K tiles async-staged to LDS; V staged transposed through
// registers; double-buffered over 64-key blocks.
// ---------------------------------------------------------------------------
__global__ __launch_bounds__(256)
void attn_kernel(const _Float16* __restrict__ Qh, const _Float16* __restrict__ Kh,
                 const _Float16* __restrict__ Vh, _Float16* __restrict__ ctx) {
  const int b = blockIdx.z, h = blockIdx.y, qt = blockIdx.x;
  const int q0 = qt * 128;
  const int tid = threadIdx.x, w = tid >> 5, L = tid & 31;
  const size_t headbase = ((size_t)b * H_ + h) * S_ * HD_;

  __shared__ alignas(16) _Float16 Qs[128 * 64];
  __shared__ alignas(16) _Float16 Ks[2][64 * 64];
  __shared__ alignas(16) _Float16 VT[2][64 * 64];   // [hd][key]
  __shared__ alignas(16) _Float16 Ps[8][16 * 64];   // per-wave P tile

  const _Float16* Qg = Qh + headbase + (size_t)q0 * HD_ + tid * 32;
  const _Float16* Kg = Kh + headbase + tid * 16;
  const int vkey = tid >> 2, vseg = (tid & 3) * 16;
  const _Float16* Vg = Vh + headbase + (size_t)vkey * HD_ + vseg;

  // stage Q tile (8192 halves, 32/thread) + first K tile, async when possible
#pragma unroll
  for (int i = 0; i < 4; ++i) async_cp16(Qg + i * 8, &Qs[tid * 32 + i * 8]);
  async_cp16(Kg,     &Ks[0][tid * 16]);
  async_cp16(Kg + 8, &Ks[0][tid * 16 + 8]);
  uint4 vr0 = ((const uint4*)Vg)[0];
  uint4 vr1 = ((const uint4*)Vg)[1];
  {
    _Float16 tmp[16];
    *(uint4*)&tmp[0] = vr0; *(uint4*)&tmp[8] = vr1;
#pragma unroll
    for (int j = 0; j < 16; ++j) VT[0][(vseg + j) * 64 + vkey] = tmp[j];
  }
  async_join();
  __syncthreads();

  float m_i[8], l_i[8];
  v8f accO[4];
#pragma unroll
  for (int vv = 0; vv < 8; ++vv) { m_i[vv] = -1.0e30f; l_i[vv] = 0.f; }
#pragma unroll
  for (int nt = 0; nt < 4; ++nt) accO[nt] = zerov8f();

  const int koff = (L >> 4) * 8;
  const int NKB = S_ / 64;

  for (int kb = 0; kb < NKB; ++kb) {
    const int cur = kb & 1;
    if (kb < NKB - 1) {  // launch next key-block staging before compute
      const size_t nb = (size_t)(kb + 1) * 64 * HD_;
      async_cp16(Kg + nb,     &Ks[cur ^ 1][tid * 16]);
      async_cp16(Kg + nb + 8, &Ks[cur ^ 1][tid * 16 + 8]);
      vr0 = ((const uint4*)(Vg + nb))[0];
      vr1 = ((const uint4*)(Vg + nb))[1];
    }

    // scores: [16q x 64key] per wave, K-dim = HD (2 WMMA k-steps)
    v8f sc[4];
#pragma unroll
    for (int nt = 0; nt < 4; ++nt) sc[nt] = zerov8f();
#pragma unroll
    for (int ks = 0; ks < 2; ++ks) {
      v16h aq  = ldfrag(&Qs[(16 * w + (L & 15)) * 64], koff + 32 * ks);
      v16h bk0 = ldfrag(&Ks[cur][(L & 15) * 64],        koff + 32 * ks);
      v16h bk1 = ldfrag(&Ks[cur][(16 + (L & 15)) * 64], koff + 32 * ks);
      v16h bk2 = ldfrag(&Ks[cur][(32 + (L & 15)) * 64], koff + 32 * ks);
      v16h bk3 = ldfrag(&Ks[cur][(48 + (L & 15)) * 64], koff + 32 * ks);
      sc[0] = wmma_f16(aq, bk0, sc[0]);
      sc[1] = wmma_f16(aq, bk1, sc[1]);
      sc[2] = wmma_f16(aq, bk2, sc[2]);
      sc[3] = wmma_f16(aq, bk3, sc[3]);
    }

    // online softmax; row r = 8*(L>>4)+vv spans one 16-lane half
#pragma unroll
    for (int vv = 0; vv < 8; ++vv) {
      float mx = fmaxf(fmaxf(sc[0][vv], sc[1][vv]), fmaxf(sc[2][vv], sc[3][vv]));
#pragma unroll
      for (int off = 8; off >= 1; off >>= 1) mx = fmaxf(mx, __shfl_xor(mx, off, 32));
      const float mnew = fmaxf(m_i[vv], mx);
      const float alpha = __expf(m_i[vv] - mnew);
      float p[4], rs = 0.f;
#pragma unroll
      for (int nt = 0; nt < 4; ++nt) { p[nt] = __expf(sc[nt][vv] - mnew); rs += p[nt]; }
#pragma unroll
      for (int off = 8; off >= 1; off >>= 1) rs += __shfl_xor(rs, off, 32);
      l_i[vv] = l_i[vv] * alpha + rs;
      m_i[vv] = mnew;
#pragma unroll
      for (int nt = 0; nt < 4; ++nt) {
        accO[nt][vv] *= alpha;
        Ps[w][(((L >> 4) * 8) + vv) * 64 + nt * 16 + (L & 15)] = (_Float16)p[nt];
      }
    }

    // O += P * V : K-dim = 64 keys (2 WMMA k-steps); per-wave Ps RAW is
    // in-order DS within the wave (compiler inserts dscnt waits)
#pragma unroll
    for (int ks = 0; ks < 2; ++ks) {
      v16h ap  = ldfrag(&Ps[w][(L & 15) * 64], koff + 32 * ks);
      v16h bv0 = ldfrag(&VT[cur][(L & 15) * 64],        koff + 32 * ks);
      v16h bv1 = ldfrag(&VT[cur][(16 + (L & 15)) * 64], koff + 32 * ks);
      v16h bv2 = ldfrag(&VT[cur][(32 + (L & 15)) * 64], koff + 32 * ks);
      v16h bv3 = ldfrag(&VT[cur][(48 + (L & 15)) * 64], koff + 32 * ks);
      accO[0] = wmma_f16(ap, bv0, accO[0]);
      accO[1] = wmma_f16(ap, bv1, accO[1]);
      accO[2] = wmma_f16(ap, bv2, accO[2]);
      accO[3] = wmma_f16(ap, bv3, accO[3]);
    }

    if (kb < NKB - 1) {  // finish V transpose staging into the other buffer
      _Float16 tmp[16];
      *(uint4*)&tmp[0] = vr0; *(uint4*)&tmp[8] = vr1;
#pragma unroll
      for (int j = 0; j < 16; ++j) VT[cur ^ 1][(vseg + j) * 64 + vkey] = tmp[j];
    }
    async_join();
    __syncthreads();
  }

  // normalize and write ctx as f16 [B,S,D]
#pragma unroll
  for (int vv = 0; vv < 8; ++vv) {
    const float inv = 1.0f / l_i[vv];
    const int srow = q0 + 16 * w + ((L >> 4) * 8) + vv;
#pragma unroll
    for (int nt = 0; nt < 4; ++nt) {
      const int d = h * HD_ + nt * 16 + (L & 15);
      ctx[((size_t)b * S_ + srow) * D_ + d] = (_Float16)(accO[nt][vv] * inv);
    }
  }
}

// ---------------------------------------------------------------------------
// Kernel 3: out = ctx(f16) @ Wo^T + bo, f32 output. A tile async-staged
// (pure f16 copy), W tile converted through registers; double-buffered.
// ---------------------------------------------------------------------------
__global__ __launch_bounds__(256)
void out_proj_kernel(const _Float16* __restrict__ A, const float* __restrict__ W,
                     const float* __restrict__ bias, float* __restrict__ out) {
  const int m0 = blockIdx.y * 128;
  const int n0 = blockIdx.x * 64;
  const int tid = threadIdx.x;
  const int w = tid >> 5, L = tid & 31;

  __shared__ alignas(16) _Float16 As[2][128][32];
  __shared__ alignas(16) _Float16 Ws[2][64][32];

  v8f acc[4];
#pragma unroll
  for (int nt = 0; nt < 4; ++nt) acc[nt] = zerov8f();

  const int ar = tid >> 1, ac = (tid & 1) * 16;
  const int wr = tid >> 2, wc = (tid & 3) * 8;
  const int koff = (L >> 4) * 8;
  const int arow = 16 * w + (L & 15);
  const int cin = L & 15;

  const _Float16* aptr = A + (size_t)(m0 + ar) * D_ + ac;
  const float*    wptr = W + (size_t)(n0 + wr) * D_ + wc;

  float4 rw0, rw1;
  async_cp16(aptr,     &As[0][ar][ac]);
  async_cp16(aptr + 8, &As[0][ar][ac + 8]);
  rw0 = ((const float4*)wptr)[0];
  rw1 = ((const float4*)wptr)[1];
  *(v8h*)&Ws[0][wr][wc] = cvt2h(rw0, rw1);
  async_join();
  __syncthreads();

  const int NIT = D_ / 32;
  for (int it = 0; it < NIT; ++it) {
    const int cur = it & 1;
    if (it < NIT - 1) {
      const _Float16* ap = aptr + (it + 1) * 32;
      const float*    wp = wptr + (it + 1) * 32;
      async_cp16(ap,     &As[cur ^ 1][ar][ac]);
      async_cp16(ap + 8, &As[cur ^ 1][ar][ac + 8]);
      rw0 = ((const float4*)wp)[0];
      rw1 = ((const float4*)wp)[1];
      if (it < NIT - 2) {
        __builtin_prefetch(ap + 32, 0, 1);
        __builtin_prefetch(wp + 32, 0, 1);
      }
    }

    v16h a  = ldfrag(&As[cur][arow][0], koff);
    v16h b0 = ldfrag(&Ws[cur][cin][0], koff);
    v16h b1 = ldfrag(&Ws[cur][16 + cin][0], koff);
    v16h b2 = ldfrag(&Ws[cur][32 + cin][0], koff);
    v16h b3 = ldfrag(&Ws[cur][48 + cin][0], koff);
    acc[0] = wmma_f16(a, b0, acc[0]);
    acc[1] = wmma_f16(a, b1, acc[1]);
    acc[2] = wmma_f16(a, b2, acc[2]);
    acc[3] = wmma_f16(a, b3, acc[3]);

    if (it < NIT - 1) {
      *(v8h*)&Ws[cur ^ 1][wr][wc] = cvt2h(rw0, rw1);
    }
    async_join();
    __syncthreads();
  }

  const int rbase = 16 * w + ((L >> 4) * 8);
#pragma unroll
  for (int nt = 0; nt < 4; ++nt) {
    const int n = n0 + nt * 16 + cin;
    const float bn = bias[n];
#pragma unroll
    for (int vv = 0; vv < 8; ++vv) {
      const int m = m0 + rbase + vv;
      out[(size_t)m * D_ + n] = acc[nt][vv] + bn;
    }
  }
}

extern "C" void kernel_launch(void* const* d_in, const int* in_sizes, int n_in,
                              void* d_out, int out_size, void* d_ws, size_t ws_size,
                              hipStream_t stream) {
  const float* query = (const float*)d_in[0];
  const float* key   = (const float*)d_in[1];
  const float* value = (const float*)d_in[2];
  const float* Wq    = (const float*)d_in[3];
  const float* bq    = (const float*)d_in[4];
  const float* Wk    = (const float*)d_in[5];
  const float* bk    = (const float*)d_in[6];
  const float* Wv    = (const float*)d_in[7];
  const float* bv    = (const float*)d_in[8];
  const float* Wo    = (const float*)d_in[9];
  const float* bo    = (const float*)d_in[10];

  // Workspace: Qh,Kh,Vh (head-major f16) + ctx (f16 [B,S,D]) = 32 MB.
  const size_t elems = (size_t)B_ * S_ * D_;
  _Float16* Qh  = (_Float16*)d_ws;
  _Float16* Kh  = Qh + elems;
  _Float16* Vh  = Kh + elems;
  _Float16* ctx = Vh + elems;

  dim3 blk(256);
  dim3 g1(D_ / 64, (B_ * S_) / 128, 3);
  hipLaunchKernelGGL(qkv_proj_kernel, g1, blk, 0, stream,
                     query, key, value, Wq, Wk, Wv, bq, bk, bv, Qh, Kh, Vh);
  dim3 g2(S_ / 128, H_, B_);
  hipLaunchKernelGGL(attn_kernel, g2, blk, 0, stream, Qh, Kh, Vh, ctx);
  dim3 g3(D_ / 64, (B_ * S_) / 128, 1);
  hipLaunchKernelGGL(out_proj_kernel, g3, blk, 0, stream, ctx, Wo, bo,
                     (float*)d_out);
}
